// Transformer_82815559401877
// MI455X (gfx1250) — matrix operand order repaired and checked by reading, not compiled, not run on hardware
//
#include <hip/hip_runtime.h>
#include <math.h>

// ---------------------------------------------------------------------------
// GPT-2 small forward on gfx1250 (CDNA5): all GEMMs + attention via
// v_wmma_f32_16x16x32_bf16 (wave32), global->LDS staging via the async
// tensor path (global_load_async_to_lds_b128 + ASYNCcnt), double-buffered.
// QKV fused into one GEMM (N=2304); attention emits bf16 directly.
// ---------------------------------------------------------------------------

typedef __bf16 v16bf __attribute__((ext_vector_type(16)));
typedef float  v8f   __attribute__((ext_vector_type(8)));

#define DMODEL 768
#define NHEADS 12
#define DHEAD  64
#define TSEQ   1024
#define BATCH  2
#define MTOK   (BATCH * TSEQ)   // 2048
#define FFDIM  3072
#define VOCAB  50257
#define VPAD   50304             // round_up(VOCAB, 128)
#define QKVN   (3 * DMODEL)      // 2304, multiple of 128

__device__ __forceinline__ unsigned short f2bf(float f) {
  unsigned int u = __float_as_uint(f);
  unsigned int r = u + 0x7fffu + ((u >> 16) & 1u);   // round-to-nearest-even
  return (unsigned short)(r >> 16);
}

union Frag16 { v16bf v; uint4 u[2]; };

// --- CDNA5 async global->LDS (ASYNCcnt-tracked) ----------------------------
__device__ __forceinline__ void async_load_b128(unsigned lds_off,
                                                unsigned long long gaddr) {
  asm volatile("global_load_async_to_lds_b128 %0, %1, off"
               :: "v"(lds_off), "v"(gaddr) : "memory");
}
__device__ __forceinline__ void wait_async0() {
#if __has_builtin(__builtin_amdgcn_s_wait_asynccnt)
  __builtin_amdgcn_s_wait_asynccnt(0);
#else
  asm volatile("s_wait_asynccnt 0x0" ::: "memory");
#endif
}

// ---------------------------------------------------------------------------
// Embedding: e[m,:] = wte[x[m],:] + wpe[m % T,:]
// ---------------------------------------------------------------------------
__global__ void embed_kernel(const int* __restrict__ x,
                             const float* __restrict__ wte,
                             const float* __restrict__ wpe,
                             float* __restrict__ e) {
  int m = blockIdx.x;
  int tok = x[m];
  int t = m % TSEQ;
  const float* src = wte + (size_t)tok * DMODEL;
  const float* pos = wpe + (size_t)t * DMODEL;
  float* dst = e + (size_t)m * DMODEL;
  for (int d = threadIdx.x; d < DMODEL; d += blockDim.x) dst[d] = src[d] + pos[d];
}

// ---------------------------------------------------------------------------
// LayerNorm over D=768, writes bf16 (feeds next GEMM). One block per row.
// ---------------------------------------------------------------------------
__global__ __launch_bounds__(256) void layernorm_bf16_kernel(
    const float* __restrict__ e, const float* __restrict__ g,
    const float* __restrict__ b, unsigned short* __restrict__ out) {
  int m = blockIdx.x;
  const float* row = e + (size_t)m * DMODEL;
  float vals[3];
  float s = 0.f, sq = 0.f;
#pragma unroll
  for (int i = 0; i < 3; i++) {
    float v = row[threadIdx.x + i * 256];
    vals[i] = v; s += v; sq += v * v;
  }
#pragma unroll
  for (int off = 16; off >= 1; off >>= 1) {
    s  += __shfl_xor(s, off, 32);
    sq += __shfl_xor(sq, off, 32);
  }
  __shared__ float ss[8], ssq[8];
  int wid = threadIdx.x >> 5, lane = threadIdx.x & 31;
  if (lane == 0) { ss[wid] = s; ssq[wid] = sq; }
  __syncthreads();
  if (wid == 0) {
    float a = (lane < 8) ? ss[lane] : 0.f;
    float c = (lane < 8) ? ssq[lane] : 0.f;
#pragma unroll
    for (int off = 4; off >= 1; off >>= 1) {
      a += __shfl_xor(a, off, 32);
      c += __shfl_xor(c, off, 32);
    }
    if (lane == 0) { ss[0] = a; ssq[0] = c; }
  }
  __syncthreads();
  float mean = ss[0] * (1.0f / DMODEL);
  float var  = ssq[0] * (1.0f / DMODEL) - mean * mean;
  float rstd = rsqrtf(var + 1e-5f);
  unsigned short* o = out + (size_t)m * DMODEL;
#pragma unroll
  for (int i = 0; i < 3; i++) {
    int d = threadIdx.x + i * 256;
    o[d] = f2bf((vals[i] - mean) * rstd * g[d] + b[d]);
  }
}

// ---------------------------------------------------------------------------
// Weight transpose+convert, LDS-tiled (coalesced both sides):
// W[K,N] f32 -> Wt[Npad,K] bf16 (zero padded rows).  Grid: (K/32, Npad/32).
// ---------------------------------------------------------------------------
__global__ __launch_bounds__(256) void wtrans_kernel(
    const float* __restrict__ W, unsigned short* __restrict__ Wt,
    int K, int N, int Npad) {
  (void)Npad;
  __shared__ float tile[32][33];
  int k0 = blockIdx.x * 32;
  int n0 = blockIdx.y * 32;
  int tx = threadIdx.x & 31, ty = threadIdx.x >> 5;   // 32 x 8
#pragma unroll
  for (int j = 0; j < 4; j++) {
    int k = ty + j * 8;
    int n = n0 + tx;
    tile[k][tx] = (n < N) ? W[(size_t)(k0 + k) * N + n] : 0.0f;
  }
  __syncthreads();
#pragma unroll
  for (int j = 0; j < 4; j++) {
    int n = ty + j * 8;
    Wt[(size_t)(n0 + n) * K + k0 + tx] = f2bf(tile[tx][n]);
  }
}

__global__ void biaspad_kernel(const float* __restrict__ b,
                               float* __restrict__ bp, int N, int Npad) {
  int i = blockIdx.x * blockDim.x + threadIdx.x;
  if (i < Npad) bp[i] = (i < N) ? b[i] : 0.0f;
}

// ---------------------------------------------------------------------------
// GEMM: C[M,N] = A[M,K](bf16) * Wt[Npad,K](bf16)^T + bias (+ GELU) (+ R)
// Block: 256 threads = 8 waves, tile 128(M) x 128(N), BK=32.
// Wave (wm 0..3, wn 0..1) computes 32x64 = 2x4 wmma accumulators.
// Global->LDS staging: async loads, double-buffered, loop-carried pointers
// (addresses advance by 64B per K-step -- no per-issue address rebuild).
// ---------------------------------------------------------------------------
#define AST 56   // LDS row stride (elements): 112B, 16B aligned, conflict-free
#define BST 56

__global__ __launch_bounds__(256) void gemm_bf16_kernel(
    const unsigned short* __restrict__ A,    // [M,K] bf16
    const unsigned short* __restrict__ Wt,   // [Npad,K] bf16
    const float* __restrict__ biasP,         // [Npad]
    float* __restrict__ Cf,                  // [M,N] f32 or null
    unsigned short* __restrict__ Cbf,        // [M,N] bf16 or null
    const float* __restrict__ R,             // [M,N] residual or null
    int M, int N, int Npad, int K, int actGelu) {
  (void)M; (void)Npad;
  __shared__ __align__(16) unsigned short As[2][128 * AST];
  __shared__ __align__(16) unsigned short Bs[2][128 * BST];

  int m0 = blockIdx.x * 128;
  int n0 = blockIdx.y * 128;
  int tid = threadIdx.x;
  int wid = tid >> 5, lane = tid & 31;
  int wm = wid & 3, wn = wid >> 2;
  int lm = lane & 15, g = lane >> 4;

  v8f acc[2][4];
#pragma unroll
  for (int i = 0; i < 2; i++)
#pragma unroll
    for (int j = 0; j < 4; j++) acc[i][j] = {};

  int arow = tid >> 2, achk = tid & 3;   // 64 rows/half x 4 chunks of 8 bf16

  // loop-carried global addresses (advance 64B per K-step) + fixed LDS offsets
  unsigned long long gaA[2], gaB[2];
  unsigned laA[2][2], laB[2][2];
#pragma unroll
  for (int hf = 0; hf < 2; hf++) {
    int r = arow + hf * 64;
    gaA[hf] = (unsigned long long)(size_t)(A  + (size_t)(m0 + r) * K + achk * 8);
    gaB[hf] = (unsigned long long)(size_t)(Wt + (size_t)(n0 + r) * K + achk * 8);
#pragma unroll
    for (int b = 0; b < 2; b++) {
      laA[b][hf] = (unsigned)(size_t)&As[b][r * AST + achk * 8];
      laB[b][hf] = (unsigned)(size_t)&Bs[b][r * BST + achk * 8];
    }
  }

  // prologue: stage tile 0
#pragma unroll
  for (int hf = 0; hf < 2; hf++) {
    async_load_b128(laA[0][hf], gaA[hf]);
    async_load_b128(laB[0][hf], gaB[hf]);
    gaA[hf] += 64; gaB[hf] += 64;
  }
  wait_async0();
  __syncthreads();

  int buf = 0;
  for (int k0 = 0; k0 < K; k0 += 32) {
    if (k0 + 32 < K) {   // stream next tile while computing this one
#pragma unroll
      for (int hf = 0; hf < 2; hf++) {
        async_load_b128(laA[buf ^ 1][hf], gaA[hf]);
        async_load_b128(laB[buf ^ 1][hf], gaB[hf]);
        gaA[hf] += 64; gaB[hf] += 64;
      }
    }
    Frag16 af[2];
#pragma unroll
    for (int i = 0; i < 2; i++) {
      const unsigned short* ap = &As[buf][(wm * 32 + i * 16 + lm) * AST];
      af[i].u[0] = *(const uint4*)(ap + 8 * g);
      af[i].u[1] = *(const uint4*)(ap + 16 + 8 * g);
    }
#pragma unroll
    for (int j = 0; j < 4; j++) {
      Frag16 bf;
      const unsigned short* bp = &Bs[buf][(wn * 64 + j * 16 + lm) * BST + 8 * g];
      bf.u[0] = *(const uint4*)bp;
      bf.u[1] = *(const uint4*)(bp + 16);
      acc[0][j] = __builtin_amdgcn_wmma_f32_16x16x32_bf16(
          false, af[0].v, false, bf.v, (short)0, acc[0][j], false, false);
      acc[1][j] = __builtin_amdgcn_wmma_f32_16x16x32_bf16(
          false, af[1].v, false, bf.v, (short)0, acc[1][j], false, false);
    }
    wait_async0();        // next tile fully resident in LDS
    __syncthreads();      // everyone done reading current tile
    buf ^= 1;
  }

  // epilogue: bias (+gelu) (+residual), store f32 and/or bf16; n bounds-checked
#pragma unroll
  for (int j = 0; j < 4; j++) {
    int n = n0 + wn * 64 + j * 16 + lm;
    if (n < N) {
      float bias = biasP[n];
#pragma unroll
      for (int i = 0; i < 2; i++) {
#pragma unroll
        for (int r = 0; r < 8; r++) {
          int m = m0 + wm * 32 + i * 16 + 8 * g + r;
          float v = acc[i][j][r] + bias;
          if (actGelu) {
            float x = v;
            v = 0.5f * x * (1.0f + tanhf(0.7978845608028654f * (x + 0.044715f * x * x * x)));
          }
          size_t idx = (size_t)m * N + n;
          if (R)   v += R[idx];
          if (Cf)  Cf[idx] = v;
          if (Cbf) Cbf[idx] = f2bf(v);
        }
      }
    }
  }
}

// ---------------------------------------------------------------------------
// Flash-style causal attention. Block = (64 queries, head h, batch b),
// 128 threads = 4 waves x 16 queries. Keys processed in chunks of 32.
// S = QK^T and O += P*V both via WMMA; K chunk staged with async loads.
// Q/K/V live in the fused QKV activation (row stride = QKVN); output bf16.
// ---------------------------------------------------------------------------
__global__ __launch_bounds__(128) void attention_kernel(
    const unsigned short* __restrict__ Q,   // [B*T, QKVN] bf16 (+0)
    const unsigned short* __restrict__ Kk,  // (+768)
    const unsigned short* __restrict__ V,   // (+1536)
    unsigned short* __restrict__ Y,         // [B*T, 768] bf16
    int stride) {
  __shared__ __align__(16) unsigned short Ks[32 * 72];      // [key][d]
  __shared__ __align__(16) unsigned short Vt[64 * 56];      // [d][key]
  __shared__ __align__(16) unsigned short Ps[4 * 16 * 40];  // per-wave P tile

  int q0 = blockIdx.x * 64;
  int h  = blockIdx.y;
  int bb = blockIdx.z;
  int tid = threadIdx.x;
  int wid = tid >> 5, lane = tid & 31;
  int lm = lane & 15, g = lane >> 4;
  int qw = q0 + wid * 16;

  const size_t base = (size_t)bb * TSEQ * stride + (size_t)h * DHEAD;

  // preload Q A-fragments (d = 0..31 and 32..63)
  Frag16 qf[2];
  {
    const unsigned short* qrow = Q + base + (size_t)(qw + lm) * stride;
#pragma unroll
    for (int kk = 0; kk < 2; kk++) {
      qf[kk].u[0] = *(const uint4*)(qrow + kk * 32 + 8 * g);
      qf[kk].u[1] = *(const uint4*)(qrow + kk * 32 + 16 + 8 * g);
    }
  }

  v8f o[4];
#pragma unroll
  for (int nt = 0; nt < 4; nt++) o[nt] = {};
  float mrow[8], lrow[8];
#pragma unroll
  for (int r = 0; r < 8; r++) { mrow[r] = -3.0e38f; lrow[r] = 0.f; }

  int kend = q0 + 64;
  for (int kc = 0; kc < kend; kc += 32) {
    // stage K chunk [32 keys][64 d] via async loads (2 x b128 per thread)
#pragma unroll
    for (int t = 0; t < 2; t++) {
      int i = tid + t * 128;           // 256 chunks of 8 bf16
      int key = i >> 3, c = i & 7;
      async_load_b128((unsigned)(size_t)&Ks[key * 72 + c * 8],
                      (unsigned long long)(size_t)(Kk + base +
                          (size_t)(kc + key) * stride + c * 8));
    }
    // stage V chunk transposed: Vt[d][key] (coalesced global reads)
    for (int i = tid; i < 32 * 64; i += 128) {
      int d = i & 63, key = i >> 6;
      Vt[d * 56 + key] = V[base + (size_t)(kc + key) * stride + d];
    }
    wait_async0();
    __syncthreads();

    if (kc <= qw + 15) {            // wave-uniform: EXEC stays all-ones
      v8f s0 = {}, s1 = {};
#pragma unroll
      for (int kk = 0; kk < 2; kk++) {
        Frag16 b0, b1;
        const unsigned short* p0 = &Ks[(0 * 16 + lm) * 72 + kk * 32 + 8 * g];
        const unsigned short* p1 = &Ks[(1 * 16 + lm) * 72 + kk * 32 + 8 * g];
        b0.u[0] = *(const uint4*)p0; b0.u[1] = *(const uint4*)(p0 + 16);
        b1.u[0] = *(const uint4*)p1; b1.u[1] = *(const uint4*)(p1 + 16);
        s0 = __builtin_amdgcn_wmma_f32_16x16x32_bf16(false, qf[kk].v, false, b0.v,
                                                     (short)0, s0, false, false);
        s1 = __builtin_amdgcn_wmma_f32_16x16x32_bf16(false, qf[kk].v, false, b1.v,
                                                     (short)0, s1, false, false);
      }
      const float scl = 0.125f;     // 1/sqrt(64)
#pragma unroll
      for (int r = 0; r < 8; r++) {
        int qrow = qw + 8 * g + r;
        float a0 = s0[r] * scl, a1 = s1[r] * scl;
        if (kc + lm > qrow)      a0 = -3.0e38f;   // causal mask
        if (kc + 16 + lm > qrow) a1 = -3.0e38f;
        float rmax = fmaxf(a0, a1);
#pragma unroll
        for (int off = 8; off >= 1; off >>= 1)
          rmax = fmaxf(rmax, __shfl_xor(rmax, off, 32));
        float newm = fmaxf(mrow[r], rmax);
        float resc = __expf(mrow[r] - newm);
        float p0 = __expf(a0 - newm);
        float p1 = __expf(a1 - newm);
        float rsum = p0 + p1;
#pragma unroll
        for (int off = 8; off >= 1; off >>= 1) rsum += __shfl_xor(rsum, off, 32);
        lrow[r] = lrow[r] * resc + rsum;
        mrow[r] = newm;
#pragma unroll
        for (int nt = 0; nt < 4; nt++) o[nt][r] *= resc;
        Ps[(wid * 16 + 8 * g + r) * 40 + lm]      = f2bf(p0);
        Ps[(wid * 16 + 8 * g + r) * 40 + 16 + lm] = f2bf(p1);
      }
      // P (16x32) * V (32x64)
      Frag16 pf;
      const unsigned short* pp = &Ps[(wid * 16 + lm) * 40];
      pf.u[0] = *(const uint4*)(pp + 8 * g);
      pf.u[1] = *(const uint4*)(pp + 16 + 8 * g);
#pragma unroll
      for (int nt = 0; nt < 4; nt++) {
        Frag16 vf;
        const unsigned short* vp = &Vt[(nt * 16 + lm) * 56 + 8 * g];
        vf.u[0] = *(const uint4*)vp; vf.u[1] = *(const uint4*)(vp + 16);
        o[nt] = __builtin_amdgcn_wmma_f32_16x16x32_bf16(
            false, pf.v, false, vf.v, (short)0, o[nt], false, false);
      }
    }
    __syncthreads();
  }

#pragma unroll
  for (int r = 0; r < 8; r++) {
    float inv = 1.0f / lrow[r];
    int qrow = qw + 8 * g + r;
    unsigned short* dst =
        Y + (size_t)bb * TSEQ * DMODEL + (size_t)qrow * DMODEL + (size_t)h * DHEAD;
#pragma unroll
    for (int nt = 0; nt < 4; nt++) dst[nt * 16 + lm] = f2bf(o[nt][r] * inv);
  }
}

// ---------------------------------------------------------------------------
// Host orchestration
// ---------------------------------------------------------------------------
static void run_linear(const float* W, const float* bias, int K, int N, int Npad,
                       const unsigned short* Abf, float* Cf, unsigned short* Cbf,
                       const float* R, int actGelu,
                       unsigned short* Wt, float* bp, hipStream_t stream) {
  dim3 tgrid(K / 32, Npad / 32);
  wtrans_kernel<<<tgrid, 256, 0, stream>>>(W, Wt, K, N, Npad);
  biaspad_kernel<<<(Npad + 255) / 256, 256, 0, stream>>>(bias, bp, N, Npad);
  dim3 grid(MTOK / 128, Npad / 128);
  gemm_bf16_kernel<<<grid, 256, 0, stream>>>(Abf, Wt, bp, Cf, Cbf, R,
                                             MTOK, N, Npad, K, actGelu);
}

extern "C" void kernel_launch(void* const* d_in, const int* in_sizes, int n_in,
                              void* d_out, int out_size, void* d_ws, size_t ws_size,
                              hipStream_t stream) {
  (void)in_sizes; (void)n_in; (void)out_size; (void)ws_size;
  // setup_inputs() insertion order:
  //  0: x  1: wte  2: wpe
  //  per layer L (base 3+16L): ln1_g, ln1_b, ln2_g, ln2_b,
  //      q.w, q.b, k.w, k.b, v.w, v.b, o.w, o.b, fc.w, fc.b, proj.w, proj.b
  //  67: lnf_g  68: lnf_b  69: unemb.w  70: unemb.b
  const int*   x   = (const int*)d_in[0];
  const float* wte = (const float*)d_in[1];
  const float* wpe = (const float*)d_in[2];
  const float* lnf_g   = (const float*)d_in[3 + 4 * 16 + 0];
  const float* lnf_b   = (const float*)d_in[3 + 4 * 16 + 1];
  const float* unemb_w = (const float*)d_in[3 + 4 * 16 + 2];
  const float* unemb_b = (const float*)d_in[3 + 4 * 16 + 3];

  char* ws = (char*)d_ws;
  size_t off = 0;
  auto alloc = [&](size_t bytes) -> char* {
    char* p = ws + off;
    off += (bytes + 255) & ~(size_t)255;
    return p;
  };
  float*          e    = (float*)alloc((size_t)MTOK * DMODEL * 4);
  unsigned short* hbf  = (unsigned short*)alloc((size_t)MTOK * DMODEL * 2);
  unsigned short* qkv  = (unsigned short*)alloc((size_t)MTOK * QKVN * 2);
  unsigned short* ybf  = (unsigned short*)alloc((size_t)MTOK * DMODEL * 2);
  unsigned short* gbf  = (unsigned short*)alloc((size_t)MTOK * FFDIM * 2);
  unsigned short* Wt   = (unsigned short*)alloc((size_t)VPAD * DMODEL * 2);
  float*          bp   = (float*)alloc((size_t)VPAD * 4);

  embed_kernel<<<MTOK, 256, 0, stream>>>(x, wte, wpe, e);

  for (int L = 0; L < 4; L++) {
    const float** p = (const float**)(d_in + 3 + L * 16);
    const float *ln1g = p[0], *ln1b = p[1], *ln2g = p[2], *ln2b = p[3];
    const float *qw = p[4],  *qb = p[5],  *kw = p[6],  *kb2 = p[7];
    const float *vw = p[8],  *vb = p[9],  *ow = p[10], *ob = p[11];
    const float *fcw = p[12], *fcb = p[13], *pjw = p[14], *pjb = p[15];

    layernorm_bf16_kernel<<<MTOK, 256, 0, stream>>>(e, ln1g, ln1b, hbf);

    // fused QKV: pack transposed weights/biases into row segments of Wt/bp,
    // then one GEMM with N = 2304
    {
      dim3 tg(DMODEL / 32, DMODEL / 32);
      wtrans_kernel<<<tg, 256, 0, stream>>>(qw, Wt + (size_t)0 * DMODEL * DMODEL,
                                            DMODEL, DMODEL, DMODEL);
      wtrans_kernel<<<tg, 256, 0, stream>>>(kw, Wt + (size_t)1 * DMODEL * DMODEL,
                                            DMODEL, DMODEL, DMODEL);
      wtrans_kernel<<<tg, 256, 0, stream>>>(vw, Wt + (size_t)2 * DMODEL * DMODEL,
                                            DMODEL, DMODEL, DMODEL);
      biaspad_kernel<<<3, 256, 0, stream>>>(qb,  bp + 0 * DMODEL, DMODEL, DMODEL);
      biaspad_kernel<<<3, 256, 0, stream>>>(kb2, bp + 1 * DMODEL, DMODEL, DMODEL);
      biaspad_kernel<<<3, 256, 0, stream>>>(vb,  bp + 2 * DMODEL, DMODEL, DMODEL);
      dim3 grid(MTOK / 128, QKVN / 128);
      gemm_bf16_kernel<<<grid, 256, 0, stream>>>(hbf, Wt, bp, nullptr, qkv,
                                                 nullptr, MTOK, QKVN, QKVN,
                                                 DMODEL, 0);
    }

    attention_kernel<<<dim3(TSEQ / 64, NHEADS, BATCH), 128, 0, stream>>>(
        qkv, qkv + DMODEL, qkv + 2 * DMODEL, ybf, QKVN);

    // e = e + y @ o.w + o.b
    run_linear(ow, ob, DMODEL, DMODEL, DMODEL, ybf, e, nullptr, e, 0, Wt, bp, stream);

    layernorm_bf16_kernel<<<MTOK, 256, 0, stream>>>(e, ln2g, ln2b, hbf);
    // gelu(h @ fc.w + fc.b) -> bf16
    run_linear(fcw, fcb, DMODEL, FFDIM, FFDIM, hbf, nullptr, gbf, nullptr, 1, Wt, bp, stream);
    // e = e + g @ proj.w + proj.b
    run_linear(pjw, pjb, FFDIM, DMODEL, DMODEL, gbf, e, nullptr, e, 0, Wt, bp, stream);
  }

  layernorm_bf16_kernel<<<MTOK, 256, 0, stream>>>(e, lnf_g, lnf_b, hbf);
  // logits = h @ unemb.w + unemb.b  (dominant GEMM: 2048 x 50257 x 768)
  run_linear(unemb_w, unemb_b, DMODEL, VOCAB, VPAD, hbf, (float*)d_out, nullptr,
             nullptr, 0, Wt, bp, stream);
}
